// HydraAttention_20255065768249
// MI455X (gfx1250) — matrix operand order repaired and checked by reading, not compiled, not run on hardware
//
#include <hip/hip_runtime.h>

typedef __attribute__((ext_vector_type(16))) _Float16 v16h;
typedef __attribute__((ext_vector_type(8)))  _Float16 v8h;
typedef __attribute__((ext_vector_type(4)))  float    v4f;
typedef __attribute__((ext_vector_type(8)))  float    v8f;
typedef __attribute__((address_space(3)))    _Float16 lds_h;

#define GEMM_M 32768
#define GEMM_N 3072
#define GEMM_K 1024
#define D_DIM  1024
#define T_DIM  8192
#define LSB    40   // LDS tile row stride in halves (32 + 8 pad; 80B: b128-aligned, bank-spread)

// ---------------------------------------------------------------------------
// Async global->LDS 128-bit copy (CDNA5 Tensor/async path, ASYNCcnt-tracked)
// ---------------------------------------------------------------------------
__device__ __forceinline__ void async_b128(unsigned lds_addr, const _Float16* g)
{
    asm volatile("global_load_async_to_lds_b128 %0, %1, off"
                 :: "v"(lds_addr), "v"((unsigned long long)g)
                 : "memory");
}
__device__ __forceinline__ void wait_async_le4() {
    asm volatile("s_wait_asynccnt 0x4" ::: "memory");
}
__device__ __forceinline__ void wait_async_0() {
    asm volatile("s_wait_asynccnt 0x0" ::: "memory");
}

// ---------------------------------------------------------------------------
// Kernel 0a: X f32 -> f16 (contiguous, 8 elems/thread)
// ---------------------------------------------------------------------------
__global__ __launch_bounds__(256) void cvt_x_kernel(
    const float* __restrict__ X, _Float16* __restrict__ Xh)
{
    int i = (blockIdx.x * 256 + threadIdx.x) * 8;
    v4f a = *(const v4f*)(X + i);
    v4f b = *(const v4f*)(X + i + 4);
    v8h h;
#pragma unroll
    for (int j = 0; j < 4; ++j) { h[j] = (_Float16)a[j]; h[4 + j] = (_Float16)b[j]; }
    *(v8h*)(Xh + i) = h;
}

// ---------------------------------------------------------------------------
// Kernel 0b: W (1024x3072 row-major f32) -> Wt (3072x1024 n-major f16),
// LDS-tiled transpose, 64(n) x 32(k) tiles.
// ---------------------------------------------------------------------------
__global__ __launch_bounds__(256) void cvt_wt_kernel(
    const float* __restrict__ W, _Float16* __restrict__ Wt)
{
    __shared__ _Float16 tile[32][72];            // 64 + 8 pad
    int n0 = blockIdx.x * 64;                    // 48 blocks
    int k0 = blockIdx.y * 32;                    // 32 blocks
#pragma unroll
    for (int i = 0; i < 8; ++i) {
        int e  = threadIdx.x + i * 256;          // 0..2047
        int kk = e >> 6;
        int nn = e & 63;
        tile[kk][nn] = (_Float16)W[(k0 + kk) * GEMM_N + n0 + nn];
    }
    __syncthreads();
    int nn = threadIdx.x >> 2;                   // 0..63
    int kc = (threadIdx.x & 3) * 8;              // 0,8,16,24
    v8h o;
#pragma unroll
    for (int j = 0; j < 8; ++j) o[j] = tile[kc + j][nn];
    *(v8h*)(Wt + (size_t)(n0 + nn) * GEMM_K + k0 + kc) = o;
}

// ---------------------------------------------------------------------------
// WMMA compute for one staged K-step (both tiles f16, K-contiguous rows)
// ---------------------------------------------------------------------------
__device__ __forceinline__ void wmma_stage(
    const _Float16* __restrict__ As, const _Float16* __restrict__ Bs,
    int wm, int wn, int lrow, int lhi, v8f acc[2][4])
{
    // A 16x32 f16: lanes 0-15 -> M=lane, K={0..7,16..23}; lanes 16-31 -> K={8..15,24..31}
    v16h af[2];
#pragma unroll
    for (int i = 0; i < 2; ++i) {
        int row = wm * 32 + i * 16 + lrow;
        int kb  = lhi ? 8 : 0;
        v8h lo = *(const v8h*)&As[row * LSB + kb];
        v8h hi = *(const v8h*)&As[row * LSB + kb + 16];
        af[i] = __builtin_shufflevector(lo, hi,
                 0,1,2,3,4,5,6,7,8,9,10,11,12,13,14,15);
    }
    // B 32x16 f16: lanes 0-15 -> N=lane, K={0..7,16..23}; lanes 16-31 -> K={8..15,24..31}
    v16h bf[4];
#pragma unroll
    for (int j = 0; j < 4; ++j) {
        int n  = wn * 64 + j * 16 + lrow;
        int kb = lhi ? 8 : 0;
        v8h lo = *(const v8h*)&Bs[n * LSB + kb];
        v8h hi = *(const v8h*)&Bs[n * LSB + kb + 16];
        bf[j] = __builtin_shufflevector(lo, hi,
                 0,1,2,3,4,5,6,7,8,9,10,11,12,13,14,15);
    }
#pragma unroll
    for (int i = 0; i < 2; ++i)
#pragma unroll
        for (int j = 0; j < 4; ++j)
            acc[i][j] = __builtin_amdgcn_wmma_f32_16x16x32_f16(
                false, af[i], false, bf[j], (short)0, acc[i][j], false, false);
}

// ---------------------------------------------------------------------------
// Kernel 1: qkv = Xh @ Wt^T + bqkv. 256 thr / 8 waves, 128x128 tile, K-step 32,
// double-buffered async global->LDS staging.
// ---------------------------------------------------------------------------
__global__ __launch_bounds__(256) void qkv_gemm_wmma_kernel(
    const _Float16* __restrict__ Xh,   // (32768,1024) row-major
    const _Float16* __restrict__ Wt,   // (3072,1024)  n-major
    const float* __restrict__ Bq,      // (3072)
    float* __restrict__ QKV)           // (32768,3072)
{
    __shared__ _Float16 As[2][128 * LSB];
    __shared__ _Float16 Bs[2][128 * LSB];

    const int tid  = threadIdx.x;
    const int lane = tid & 31;
    const int wave = tid >> 5;
    const int wm   = wave >> 1;          // 0..3 (M)
    const int wn   = wave & 1;           // 0..1 (N)
    const int tileM = blockIdx.y * 128;
    const int tileN = blockIdx.x * 128;
    const int lrow = lane & 15;
    const int lhi  = lane >> 4;

    // staging role: 2 threads per tile row, 32B (16 halves) each
    const int crow  = tid >> 1;          // 0..127
    const int cpart = (tid & 1) * 16;    // halves
    const _Float16* gA = Xh + (size_t)(tileM + crow) * GEMM_K + cpart;
    const _Float16* gB = Wt + (size_t)(tileN + crow) * GEMM_K + cpart;
    unsigned aoff[2], boff[2];
#pragma unroll
    for (int s = 0; s < 2; ++s) {
        aoff[s] = (unsigned)(size_t)(lds_h*)&As[s][crow * LSB + cpart];
        boff[s] = (unsigned)(size_t)(lds_h*)&Bs[s][crow * LSB + cpart];
    }

    v8f acc[2][4];
#pragma unroll
    for (int i = 0; i < 2; ++i)
#pragma unroll
        for (int j = 0; j < 4; ++j)
#pragma unroll
            for (int r = 0; r < 8; ++r) acc[i][j][r] = 0.0f;

    // prologue: stage k-step 0 into buffer 0
    async_b128(aoff[0],      gA);
    async_b128(aoff[0] + 16, gA + 8);
    async_b128(boff[0],      gB);
    async_b128(boff[0] + 16, gB + 8);

    const int NSTEP = GEMM_K / 32;       // 32
    for (int ks = 0; ks < NSTEP; ++ks) {
        const int cur = ks & 1;
        if (ks + 1 < NSTEP) {
            // buffer cur^1 was fully consumed (trailing barrier last iter)
            const int k1 = (ks + 1) * 32;
            async_b128(aoff[cur ^ 1],      gA + k1);
            async_b128(aoff[cur ^ 1] + 16, gA + k1 + 8);
            async_b128(boff[cur ^ 1],      gB + k1);
            async_b128(boff[cur ^ 1] + 16, gB + k1 + 8);
            wait_async_le4();            // in-order: current 4 have landed
        } else {
            wait_async_0();
        }
        __syncthreads();
        wmma_stage(As[cur], Bs[cur], wm, wn, lrow, lhi, acc);
        __syncthreads();
    }

    // epilogue: add bias, store f32 (lanes 0-15 / 16-31 each cover 64B segments)
#pragma unroll
    for (int j = 0; j < 4; ++j) {
        int col  = tileN + wn * 64 + j * 16 + lrow;
        float bv = Bq[col];
#pragma unroll
        for (int i = 0; i < 2; ++i) {
            int rbase = tileM + wm * 32 + i * 16 + lhi * 8;
#pragma unroll
            for (int r = 0; r < 8; ++r)
                QKV[(size_t)(rbase + r) * GEMM_N + col] = acc[i][j][r] + bv;
        }
    }
}

// ---------------------------------------------------------------------------
// Kernel 2: per-row inverse L2 norms of q and k
// ---------------------------------------------------------------------------
__global__ __launch_bounds__(256) void rownorm_kernel(
    const float* __restrict__ qkv, float* __restrict__ invq, float* __restrict__ invk)
{
    int row = blockIdx.x;
    const float* q = qkv + (size_t)row * GEMM_N;
    const float* k = q + D_DIM;
    float sq = 0.0f, sk = 0.0f;
    for (int i = threadIdx.x; i < D_DIM; i += 256) {
        float a = q[i]; sq += a * a;
        float b = k[i]; sk += b * b;
    }
    __shared__ float red[2][256];
    red[0][threadIdx.x] = sq;
    red[1][threadIdx.x] = sk;
    __syncthreads();
    for (int s = 128; s > 0; s >>= 1) {
        if (threadIdx.x < s) {
            red[0][threadIdx.x] += red[0][threadIdx.x + s];
            red[1][threadIdx.x] += red[1][threadIdx.x + s];
        }
        __syncthreads();
    }
    if (threadIdx.x == 0) {
        invq[row] = rsqrtf(red[0][0]);
        invk[row] = rsqrtf(red[1][0]);
    }
}

__global__ void kv_init_kernel(float* __restrict__ kv)
{
    int i = blockIdx.x * 256 + threadIdx.x;
    if (i < 4 * D_DIM) kv[i] = 0.0f;
}

// ---------------------------------------------------------------------------
// Kernel 3: kv[b,d] = sum_t khat[b,t,d]*v[b,t,d]; register acc + one atomicAdd
// ---------------------------------------------------------------------------
__global__ __launch_bounds__(256) void kv_accum_kernel(
    const float* __restrict__ qkv, const float* __restrict__ invk,
    float* __restrict__ kv)
{
    int b  = blockIdx.z;
    int d  = blockIdx.y * 256 + threadIdx.x;
    int t0 = blockIdx.x * 512;
    float acc = 0.0f;
    for (int t = t0; t < t0 + 512; ++t) {
        int row = b * T_DIM + t;
        const float* base = qkv + (size_t)row * GEMM_N;
        acc += base[D_DIM + d] * invk[row] * base[2 * D_DIM + d];
    }
    atomicAdd(&kv[b * D_DIM + d], acc);
}

// ---------------------------------------------------------------------------
// Kernel 4: out = qhat * kv * scale + bias
// ---------------------------------------------------------------------------
__global__ __launch_bounds__(256) void out_kernel(
    const float* __restrict__ qkv, const float* __restrict__ invq,
    const float* __restrict__ kv, const float* __restrict__ scale,
    const float* __restrict__ bias, float* __restrict__ out)
{
    int row = blockIdx.x;
    int b   = row >> 13;
    int d   = threadIdx.x * 4;
    float iq = invq[row];
    v4f qv  = *(const v4f*)(qkv + (size_t)row * GEMM_N + d);
    v4f kvv = *(const v4f*)(kv + b * D_DIM + d);
    v4f sc  = *(const v4f*)(scale + d);
    v4f bi  = *(const v4f*)(bias + d);
    v4f o;
#pragma unroll
    for (int j = 0; j < 4; ++j) o[j] = qv[j] * iq * kvv[j] * sc[j] + bi[j];
    *(v4f*)(out + (size_t)row * D_DIM + d) = o;
}

// ---------------------------------------------------------------------------
extern "C" void kernel_launch(void* const* d_in, const int* in_sizes, int n_in,
                              void* d_out, int out_size, void* d_ws, size_t ws_size,
                              hipStream_t stream)
{
    (void)in_sizes; (void)n_in; (void)out_size; (void)ws_size;
    const float* x     = (const float*)d_in[0];
    const float* Wqkv  = (const float*)d_in[1];
    const float* bqkv  = (const float*)d_in[2];
    const float* scale = (const float*)d_in[3];
    const float* bias  = (const float*)d_in[4];
    float* out = (float*)d_out;

    float* qkv  = (float*)d_ws;                              // 32768*3072 f32
    float* invq = qkv + (size_t)GEMM_M * GEMM_N;             // 32768
    float* invk = invq + GEMM_M;                             // 32768
    float* kv   = invk + GEMM_M;                             // 4096
    _Float16* Xh = (_Float16*)(kv + 4 * D_DIM);              // 32768*1024 f16
    _Float16* Wt = Xh + (size_t)GEMM_M * GEMM_K;             // 3072*1024 f16

    cvt_x_kernel<<<GEMM_M * GEMM_K / (256 * 8), 256, 0, stream>>>(x, Xh);
    cvt_wt_kernel<<<dim3(GEMM_N / 64, GEMM_K / 32), 256, 0, stream>>>(Wqkv, Wt);
    qkv_gemm_wmma_kernel<<<dim3(GEMM_N / 128, GEMM_M / 128), 256, 0, stream>>>(
        Xh, Wt, bqkv, qkv);
    rownorm_kernel<<<GEMM_M, 256, 0, stream>>>(qkv, invq, invk);
    kv_init_kernel<<<16, 256, 0, stream>>>(kv);
    kv_accum_kernel<<<dim3(16, 4, 4), 256, 0, stream>>>(qkv, invk, kv);
    out_kernel<<<GEMM_M, 256, 0, stream>>>(qkv, invq, kv, scale, bias, out);
}
